// CausalSelfAttention_9500467659037
// MI455X (gfx1250) — compile-verified, matrix operand order checked by reading, not run on hardware
//
#include <hip/hip_runtime.h>
#include <math.h>

typedef __attribute__((ext_vector_type(16))) _Float16 v16h;
typedef __attribute__((ext_vector_type(8)))  _Float16 v8h;
typedef __attribute__((ext_vector_type(8)))  float    v8f;

union AFrag { v16h v; v8h h[2]; };

#define B_  2
#define T_  2048
#define C_  2048
#define H_  16
#define HD_ 128
#define M_  4096            // B*T
#define WELEM (8*256*256)   // elements per weight tensor

__constant__ int c_idx[8][8] = {
  {0,1,2,3,4,5,6,7},{1,0,3,2,5,4,7,6},{2,3,0,1,6,7,4,5},{3,2,1,0,7,6,5,4},
  {4,5,6,7,0,1,2,3},{5,4,7,6,1,0,3,2},{6,7,4,5,2,3,0,1},{7,6,5,4,3,2,1,0}};
__constant__ float c_sgn[8][8] = {
  { 1, 1, 1, 1, 1, 1, 1, 1},{ 1,-1, 1,-1, 1,-1,-1, 1},
  { 1,-1,-1, 1, 1, 1,-1,-1},{ 1, 1,-1,-1, 1,-1, 1,-1},
  { 1,-1,-1,-1,-1, 1, 1, 1},{ 1, 1,-1, 1,-1,-1,-1, 1},
  { 1, 1, 1,-1,-1, 1,-1,-1},{ 1,-1, 1, 1,-1,-1, 1,-1}};

// ---- per-weight scale s = mean(|W|)+1e-8 --------------------------------
__global__ void scale_kernel(const float* w0, const float* w1,
                             const float* w2, const float* w3, float* sbuf) {
  const float* w = (blockIdx.x==0)?w0:(blockIdx.x==1)?w1:(blockIdx.x==2)?w2:w3;
  __shared__ float red[256];
  float sum = 0.f;
  for (int i = threadIdx.x; i < WELEM; i += 256) sum += fabsf(w[i]);
  red[threadIdx.x] = sum;
  __syncthreads();
  for (int step = 128; step > 0; step >>= 1) {
    if (threadIdx.x < step) red[threadIdx.x] += red[threadIdx.x + step];
    __syncthreads();
  }
  if (threadIdx.x == 0) sbuf[blockIdx.x] = red[0] / (float)WELEM + 1e-8f;
}

// ---- build Wbig^T (f16, [N=2048][K=2048]) from ternary blocks -----------
__global__ void build_kernel(const float* w0, const float* w1,
                             const float* w2, const float* w3,
                             const float* sbuf,
                             _Float16* o0, _Float16* o1,
                             _Float16* o2, _Float16* o3) {
  int wi = blockIdx.y;
  const float* W = (wi==0)?w0:(wi==1)?w1:(wi==2)?w2:w3;
  _Float16*    O = (wi==0)?o0:(wi==1)?o1:(wi==2)?o2:o3;
  float s = sbuf[wi];
  int id  = blockIdx.x * 256 + threadIdx.x;   // < 2048*2048
  int col = id >> 11;                         // output feature
  int row = id & 2047;                        // input feature
  int kB = col >> 8, qc = col & 255;
  int jB = row >> 8, p  = row & 255;
  int i = 0;
  #pragma unroll
  for (int t = 0; t < 8; ++t) if (c_idx[t][jB] == kB) i = t;
  float sign = c_sgn[i][jB];
  float wv = W[((i*256 + p) << 8) + qc];
  float t = fminf(fmaxf(wv / s, -1.0f), 1.0f);
  O[(size_t)col * 2048 + row] = (_Float16)(sign * rintf(t) * s);
}

// ---- f32 -> f16 ----------------------------------------------------------
__global__ void cvt_kernel(const float* x, _Float16* o, int n) {
  int id = blockIdx.x * 256 + threadIdx.x;
  if (id < n) o[id] = (_Float16)x[id];
}

// ---- WMMA GEMM: C[M][N] = A[M][K] * Bt[N][K]^T, f32 accumulate ----------
// Wave tile 16x128: A-frag reused x8; all frags preloaded before the WMMA
// chain so the scheduler can issue one load clause and stagger the waits.
__global__ __launch_bounds__(256)
void gemm_wmma(const _Float16* __restrict__ A, const _Float16* __restrict__ Bt,
               _Float16* __restrict__ outH, float* __restrict__ outF,
               int M, int N, int K, int storeF32) {
  const int lane   = threadIdx.x & 31;
  const int wave   = threadIdx.x >> 5;
  const int lane16 = lane & 15;
  const int half   = lane >> 4;
  const int m0 = blockIdx.y * 128 + wave * 16;
  const int n0 = blockIdx.x * 128;
  v8f z = {0,0,0,0,0,0,0,0};
  v8f acc[8] = {z, z, z, z, z, z, z, z};
  const _Float16* ap = A  + (size_t)(m0 + lane16) * K + 8*half;
  const _Float16* bp = Bt + (size_t)(n0 + lane16) * K + 8*half;
  for (int kk = 0; kk < K; kk += 32) {
    AFrag a;
    a.h[0] = *(const v8h*)(ap);
    a.h[1] = *(const v8h*)(ap + 16);
    AFrag b[8];
    #pragma unroll
    for (int t = 0; t < 8; ++t) {
      b[t].h[0] = *(const v8h*)(bp + (size_t)t*16*K);
      b[t].h[1] = *(const v8h*)(bp + (size_t)t*16*K + 16);
    }
    if (kk + 256 < K) {
      __builtin_prefetch(ap + 256, 0, 3);
      __builtin_prefetch(bp + 256, 0, 3);
    }
    #pragma unroll
    for (int t = 0; t < 8; ++t)
      acc[t] = __builtin_amdgcn_wmma_f32_16x16x32_f16(
                 false, a.v, false, b[t].v, (short)0, acc[t], false, false);
    ap += 32;
    bp += 32;
  }
  #pragma unroll
  for (int t = 0; t < 8; ++t)
    #pragma unroll
    for (int r = 0; r < 8; ++r) {
      int m = m0 + r + 8*half;
      int n = n0 + t*16 + lane16;
      if (storeF32) outF[(size_t)m * N + n] = acc[t][r];
      else          outH[(size_t)m * N + n] = (_Float16)acc[t][r];
    }
}

// ---- RoPE in-place on f16 [B,T,H,HD] ------------------------------------
__global__ void rope_kernel(_Float16* buf, const float* fc, const float* fs) {
  int id = blockIdx.x * 256 + threadIdx.x;  // < B*T*H*64 = 2^22
  int dp = id & 63;
  int h  = (id >> 6) & 15;
  int t  = (id >> 10) & 2047;
  int b  = id >> 21;
  size_t base = ((size_t)(b*T_ + t))*C_ + h*HD_ + 2*dp;
  float c = fc[t*64 + dp], s = fs[t*64 + dp];
  float xr = (float)buf[base], xi = (float)buf[base + 1];
  buf[base]     = (_Float16)(xr*c - xi*s);
  buf[base + 1] = (_Float16)(xr*s + xi*c);
}

// ---- V [b][t][h][d] -> Vt [b][h][d][t] ----------------------------------
__global__ void vtrans_kernel(const _Float16* V, _Float16* Vt) {
  int id = blockIdx.x * 256 + threadIdx.x;  // < 2^23
  int t = id & 2047;
  int d = (id >> 11) & 127;
  int h = (id >> 18) & 15;
  int b = id >> 22;
  Vt[id] = V[((size_t)(b*T_ + t))*C_ + h*HD_ + d];
}

// ---- flash attention: 1 wave per 16 query rows --------------------------
__global__ __launch_bounds__(128)
void attn_kernel(const _Float16* __restrict__ Q, const _Float16* __restrict__ K,
                 const _Float16* __restrict__ Vt, _Float16* __restrict__ Y) {
  __shared__ alignas(16) _Float16 pbuf[4][16][32];
  const int lane   = threadIdx.x & 31;
  const int w      = threadIdx.x >> 5;
  const int lane16 = lane & 15;
  const int half   = lane >> 4;
  int g  = blockIdx.x * 4 + w;
  int qt = g & 127;
  int bh = g >> 7;
  int b  = bh >> 4;
  int h  = bh & 15;
  int qb = qt * 16;
  const float scale = 0.08838834764831845f;  // 1/sqrt(128)

  AFrag qa[4];
  {
    const _Float16* qp = Q + ((size_t)(b*T_ + qb + lane16))*C_ + h*HD_;
    #pragma unroll
    for (int c = 0; c < 4; ++c) {
      qa[c].h[0] = *(const v8h*)(qp + c*32 + 8*half);
      qa[c].h[1] = *(const v8h*)(qp + c*32 + 16 + 8*half);
    }
  }
  v8f z = {0,0,0,0,0,0,0,0};
  v8f acc[8];
  float mr[8], l[8];
  #pragma unroll
  for (int d = 0; d < 8; ++d) { acc[d] = z; mr[d] = -3.0e38f; l[d] = 0.f; }

  for (int s = 0; s < qb + 16; s += 32) {
    v8f s0 = z, s1 = z;
    #pragma unroll
    for (int c = 0; c < 4; ++c) {
      AFrag kb0, kb1;
      const _Float16* kp0 = K + ((size_t)(b*T_ + s + lane16))*C_ + h*HD_ + c*32;
      const _Float16* kp1 = K + ((size_t)(b*T_ + s + 16 + lane16))*C_ + h*HD_ + c*32;
      kb0.h[0] = *(const v8h*)(kp0 + 8*half);
      kb0.h[1] = *(const v8h*)(kp0 + 16 + 8*half);
      kb1.h[0] = *(const v8h*)(kp1 + 8*half);
      kb1.h[1] = *(const v8h*)(kp1 + 16 + 8*half);
      s0 = __builtin_amdgcn_wmma_f32_16x16x32_f16(false, qa[c].v, false, kb0.v,
                                                  (short)0, s0, false, false);
      s1 = __builtin_amdgcn_wmma_f32_16x16x32_f16(false, qa[c].v, false, kb1.v,
                                                  (short)0, s1, false, false);
    }
    // online softmax (rows stripe over VGPR index + lane-half; cols over lanes)
    #pragma unroll
    for (int r = 0; r < 8; ++r) {
      int row = qb + r + 8*half;
      float a0 = s0[r]*scale, a1 = s1[r]*scale;
      if (s + lane16 > row)      a0 = -3.0e38f;
      if (s + 16 + lane16 > row) a1 = -3.0e38f;
      float t = fmaxf(a0, a1);
      t = fmaxf(t, __shfl_xor(t, 1, 32));
      t = fmaxf(t, __shfl_xor(t, 2, 32));
      t = fmaxf(t, __shfl_xor(t, 4, 32));
      t = fmaxf(t, __shfl_xor(t, 8, 32));
      float mn = fmaxf(mr[r], t);
      float alpha = __expf(mr[r] - mn);
      float p0 = __expf(a0 - mn);
      float p1 = __expf(a1 - mn);
      float rs = p0 + p1;
      rs += __shfl_xor(rs, 1, 32);
      rs += __shfl_xor(rs, 2, 32);
      rs += __shfl_xor(rs, 4, 32);
      rs += __shfl_xor(rs, 8, 32);
      l[r]  = l[r]*alpha + rs;
      mr[r] = mn;
      #pragma unroll
      for (int d = 0; d < 8; ++d) acc[d][r] *= alpha;
      pbuf[w][r + 8*half][lane16]      = (_Float16)p0;
      pbuf[w][r + 8*half][16 + lane16] = (_Float16)p1;
    }
    asm volatile("s_wait_dscnt 0" ::: "memory");  // C-layout -> A-layout via LDS
    AFrag pa;
    pa.h[0] = *(const v8h*)&pbuf[w][lane16][8*half];
    pa.h[1] = *(const v8h*)&pbuf[w][lane16][16 + 8*half];
    #pragma unroll
    for (int d = 0; d < 8; ++d) {
      AFrag vb;
      const _Float16* vp = Vt + ((size_t)(bh*HD_ + d*16 + lane16))*T_ + s;
      vb.h[0] = *(const v8h*)(vp + 8*half);
      vb.h[1] = *(const v8h*)(vp + 16 + 8*half);
      acc[d] = __builtin_amdgcn_wmma_f32_16x16x32_f16(false, pa.v, false, vb.v,
                                                      (short)0, acc[d], false, false);
    }
  }
  #pragma unroll
  for (int r = 0; r < 8; ++r) {
    float inv = 1.0f / l[r];
    _Float16* yp = Y + ((size_t)(b*T_ + qb + r + 8*half))*C_ + h*HD_;
    #pragma unroll
    for (int d = 0; d < 8; ++d)
      yp[d*16 + lane16] = (_Float16)(acc[d][r] * inv);
  }
}

extern "C" void kernel_launch(void* const* d_in, const int* in_sizes, int n_in,
                              void* d_out, int out_size, void* d_ws, size_t ws_size,
                              hipStream_t stream) {
  const float* x  = (const float*)d_in[0];
  const float* wq = (const float*)d_in[1];
  const float* wk = (const float*)d_in[2];
  const float* wv = (const float*)d_in[3];
  const float* wo = (const float*)d_in[4];
  const float* fc = (const float*)d_in[5];
  const float* fs = (const float*)d_in[6];
  float* out = (float*)d_out;

  char* ws = (char*)d_ws;
  const size_t WBH = (size_t)2048 * 2048;      // elements per Wbig^T
  const size_t ACT = (size_t)M_ * C_;          // elements per activation buf
  float*    sbuf = (float*)ws;
  _Float16* wbq  = (_Float16*)(ws + 256);
  _Float16* wbk  = wbq + WBH;
  _Float16* wbv  = wbk + WBH;
  _Float16* wbo  = wbv + WBH;
  _Float16* xh   = wbo + WBH;
  _Float16* qh   = xh + ACT;
  _Float16* kh   = qh + ACT;
  _Float16* vh   = kh + ACT;
  _Float16* vt   = vh + ACT;
  _Float16* yh   = xh;                         // x-f16 dead after QKV GEMMs

  scale_kernel<<<4, 256, 0, stream>>>(wq, wk, wv, wo, sbuf);
  build_kernel<<<dim3(4194304/256, 4), 256, 0, stream>>>(
      wq, wk, wv, wo, sbuf, wbq, wbk, wbv, wbo);
  cvt_kernel<<<(int)(ACT/256), 256, 0, stream>>>(x, xh, (int)ACT);

  dim3 ggrid(2048/128, 4096/128);
  gemm_wmma<<<ggrid, 256, 0, stream>>>(xh, wbq, qh, nullptr, M_, C_, C_, 0);
  gemm_wmma<<<ggrid, 256, 0, stream>>>(xh, wbk, kh, nullptr, M_, C_, C_, 0);
  gemm_wmma<<<ggrid, 256, 0, stream>>>(xh, wbv, vh, nullptr, M_, C_, C_, 0);

  rope_kernel<<<4194304/256, 256, 0, stream>>>(qh, fc, fs);
  rope_kernel<<<4194304/256, 256, 0, stream>>>(kh, fc, fs);
  vtrans_kernel<<<8388608/256, 256, 0, stream>>>(vh, vt);

  attn_kernel<<<(B_*H_*(T_/16))/4, 128, 0, stream>>>(qh, kh, vt, yh);

  gemm_wmma<<<ggrid, 256, 0, stream>>>(yh, wbo, nullptr, out, M_, C_, C_, 1);
}
